// PairwiseScore_4286377362076
// MI455X (gfx1250) — matrix operand order repaired and checked by reading, not compiled, not run on hardware
//
#include <hip/hip_runtime.h>

// ---------------- CDNA5 WMMA types ----------------
typedef __attribute__((ext_vector_type(16))) __bf16 v16bf;
typedef __attribute__((ext_vector_type(8)))  float  v8f;
typedef int v4i_ __attribute__((vector_size(16)));   // matches builtin param type

#define N_   1024
#define G_   620
#define GP   640     // padded G (K dim of layer-1 cross GEMM)
#define F_   20
#define H_   150
#define HP   160     // padded H
#define K_   128

// Async global->LDS path (CDNA5): guarded so compilation never regresses.
#if __has_builtin(__builtin_amdgcn_global_load_async_to_lds_b128)
#define USE_ASYNC_LDS 1
#endif

static __device__ __forceinline__ unsigned short f2bf(float f) {
    union { float f; unsigned u; } x; x.f = f;
    unsigned r = x.u + 0x7FFFu + ((x.u >> 16) & 1u);   // round-to-nearest-even
    return (unsigned short)(r >> 16);
}

static __device__ __forceinline__ void wait_async0() {
#ifdef USE_ASYNC_LDS
#if __has_builtin(__builtin_amdgcn_s_wait_asynccnt)
    __builtin_amdgcn_s_wait_asynccnt(0);
#else
    asm volatile("s_wait_asynccnt 0" ::: "memory");
#endif
#endif
}

// Stage one 160x32 bf16 B chunk (10 KB) from global into LDS.
static __device__ __forceinline__ void stage_chunk(const unsigned short* __restrict__ src,
                                                   int stride, int c,
                                                   unsigned short* __restrict__ dst, int tid) {
#ifdef USE_ASYNC_LDS
    for (int idx = tid; idx < 640; idx += 256) {
        int row = idx >> 2, seg = idx & 3;
        __builtin_amdgcn_global_load_async_to_lds_b128(
            (v4i_*)(src + row * stride + c * 32 + seg * 8),
            (v4i_*)(dst + row * 32 + seg * 8),
            0, 0);
    }
#else
    for (int idx = tid; idx < 640; idx += 256) {
        int row = idx >> 2, seg = idx & 3;
        ((uint4*)dst)[row * 4 + seg] = *(const uint4*)(src + row * stride + c * 32 + seg * 8);
    }
#endif
}

// ---------------- prep kernels ----------------
__global__ void k_gpad(const float* __restrict__ g, float* __restrict__ gpad) {
    int idx = blockIdx.x * 256 + threadIdx.x;
    if (idx >= N_ * GP) return;
    int n = idx / GP, d = idx - n * GP;
    gpad[idx] = (d < G_) ? g[n * G_ + d] : 0.f;
}

// W1 cross block (rows 1240..1859), transposed + bf16: W1ct[h][d]
__global__ void k_w1ct(const float* __restrict__ W1, unsigned short* __restrict__ W1ct) {
    int idx = blockIdx.x * 256 + threadIdx.x;          // h*GP + d
    if (idx >= HP * GP) return;
    int h = idx / GP, d = idx - h * GP;
    float v = (h < H_ && d < G_) ? W1[(1240 + d) * H_ + h] : 0.f;
    W1ct[idx] = f2bf(v);
}

// W2 transposed + bf16: W2t[h_out][h_in]
__global__ void k_w2t(const float* __restrict__ W2, unsigned short* __restrict__ W2t) {
    int idx = blockIdx.x * 256 + threadIdx.x;          // ho*HP + hi
    if (idx >= HP * HP) return;
    int ho = idx / HP, hi = idx - ho * HP;
    float v = (ho < H_ && hi < H_) ? W2[hi * H_ + ho] : 0.f;
    W2t[idx] = f2bf(v);
}

__global__ void k_vec(const float* __restrict__ b2, const float* __restrict__ W3,
                      float* __restrict__ b2pad, float* __restrict__ w3pad) {
    int h = threadIdx.x;
    if (h < HP) {
        b2pad[h] = (h < H_) ? b2[h] : 0.f;
        w3pad[h] = (h < H_) ? W3[h] : 0.f;
    }
}

// pre_i[n][h] = b1[h] + g[n]·W1[0:620,h] ; pre_j[n][h] = g[n]·W1[620:1240,h]
__global__ void k_pre(const float* __restrict__ g, const float* __restrict__ W1,
                      const float* __restrict__ b1,
                      float* __restrict__ pre_i, float* __restrict__ pre_j) {
    int idx = blockIdx.x * 256 + threadIdx.x;          // n*HP + h
    if (idx >= N_ * HP) return;
    int n = idx / HP, h = idx - n * HP;
    if (h >= H_) { pre_i[idx] = 0.f; pre_j[idx] = 0.f; return; }
    float ai = b1[h], aj = 0.f;
    const float* gr = g + (size_t)n * G_;
    for (int d = 0; d < G_; ++d) {
        float a = gr[d];
        ai += a * W1[d * H_ + h];
        aj += a * W1[(G_ + d) * H_ + h];
    }
    pre_i[idx] = ai; pre_j[idx] = aj;
}

// de_pre[b][h] = dist_emb[b]·W1[1860:1880,h]
__global__ void k_de(const float* __restrict__ dist_emb, const float* __restrict__ W1,
                     float* __restrict__ de_pre) {
    int idx = blockIdx.x * 256 + threadIdx.x;          // b*HP + h
    if (idx >= 9 * HP) return;
    int b = idx / HP, h = idx - b * HP;
    float s = 0.f;
    if (h < H_)
        for (int f = 0; f < F_; ++f)
            s += dist_emb[b * F_ + f] * W1[(1860 + f) * H_ + h];
    de_pre[idx] = s;
}

// ---------------- fused main kernel: one block per anchor i ----------------
__launch_bounds__(256)
__global__ void k_main(const float* __restrict__ gpad,
                       const float* __restrict__ pre_i,
                       const float* __restrict__ pre_j,
                       const float* __restrict__ de_pre,
                       const unsigned short* __restrict__ W1ct,
                       const unsigned short* __restrict__ W2t,
                       const float* __restrict__ b2pad,
                       const float* __restrict__ w3pad,
                       const float* __restrict__ si,
                       const int* __restrict__ i1,
                       const int* __restrict__ i2,
                       const float* __restrict__ b3,
                       float* __restrict__ out) {
    __shared__ float          sg_i[GP];
    __shared__ float          s_pre_i[HP];
    __shared__ float          s_de[9 * HP];
    __shared__ float          s_b2[HP];
    __shared__ float          s_w3[HP];
    __shared__ int            s_j[K_];
    __shared__ int            s_bin[K_];
    __shared__ float          s_sibase[K_];
    __shared__ int            s_valid[K_];
    __shared__ unsigned short sB[2][HP * 32];  // double-buffered B chunk: [160][32] bf16
    __shared__ unsigned short sH1[K_ * HP];    // layer-1 activations, bf16

    const int n    = blockIdx.x;
    const int tid  = threadIdx.x;
    const int w    = tid >> 5;      // wave id (0..7) -> M-tile
    const int lane = tid & 31;
    const int nn   = lane & 15;
    const int half = lane >> 4;
    const int base = half * 8;      // A-fragment K sub-offset per ISA layout

    // ---- block init ----
    for (int idx = tid; idx < GP; idx += 256) sg_i[idx] = gpad[(size_t)n * GP + idx];
    for (int idx = tid; idx < HP; idx += 256) {
        s_pre_i[idx] = pre_i[(size_t)n * HP + idx];
        s_b2[idx]    = b2pad[idx];
        s_w3[idx]    = w3pad[idx];
    }
    for (int idx = tid; idx < 9 * HP; idx += 256) s_de[idx] = de_pre[idx];
    if (tid < K_) {
        int k  = tid;
        int ji = n - K_ + k;
        int vd = (ji >= 0);
        int j  = vd ? ji : 0;
        s_j[k]      = j;
        s_valid[k]  = vd;
        s_sibase[k] = si[n] + si[j];
        int dd = i2[n] - i1[j];     // searchsorted(BINS, dd, 'right')
        s_bin[k] = (dd >= 1) + (dd >= 2) + (dd >= 3) + (dd >= 4) +
                   (dd >= 8) + (dd >= 16) + (dd >= 32) + (dd >= 64);
    }
    // kick off the first B chunk while init finishes
    stage_chunk(W1ct, GP, 0, sB[0], tid);
    __syncthreads();

    const int krow = w * 16 + nn;                 // this lane's A row (pair index k)
    const int jrow = s_j[krow];
    const float4* gj4 = (const float4*)(gpad + (size_t)jrow * GP);
    const float4* gi4 = (const float4*)sg_i;

    v8f acc[10];
    const v8f vzero = {0.f, 0.f, 0.f, 0.f, 0.f, 0.f, 0.f, 0.f};
#pragma unroll
    for (int t = 0; t < 10; ++t) acc[t] = vzero;

    // ========== layer 1: cross = (g_i .* g_j) x W1c  (K = 640, 20 chunks, dbl-buffered) ==========
    for (int c = 0; c < 20; ++c) {
        const unsigned short* cur = sB[c & 1];
        wait_async0();              // our chunk-c async loads have landed
        __syncthreads();            // everyone's landed; buffer (c+1)&1 is free
        if (c + 1 < 20) stage_chunk(W1ct, GP, c + 1, sB[(c + 1) & 1], tid);

        // build A fragment in registers: a[d] = g_i[d] * g_j[d], bf16
        int o0 = (c * 32 + base) >> 2;                   // float4 index
        float4 ja = gj4[o0], jb = gj4[o0 + 1], jc = gj4[o0 + 4], jd = gj4[o0 + 5];
        float4 ia = gi4[o0], ib = gi4[o0 + 1], ic = gi4[o0 + 4], id = gi4[o0 + 5];
        union { unsigned short s[16]; v16bf v; } af;
        af.s[0]  = f2bf(ia.x * ja.x);  af.s[1]  = f2bf(ia.y * ja.y);
        af.s[2]  = f2bf(ia.z * ja.z);  af.s[3]  = f2bf(ia.w * ja.w);
        af.s[4]  = f2bf(ib.x * jb.x);  af.s[5]  = f2bf(ib.y * jb.y);
        af.s[6]  = f2bf(ib.z * jb.z);  af.s[7]  = f2bf(ib.w * jb.w);
        af.s[8]  = f2bf(ic.x * jc.x);  af.s[9]  = f2bf(ic.y * jc.y);
        af.s[10] = f2bf(ic.z * jc.z);  af.s[11] = f2bf(ic.w * jc.w);
        af.s[12] = f2bf(id.x * jd.x);  af.s[13] = f2bf(id.y * jd.y);
        af.s[14] = f2bf(id.z * jd.z);  af.s[15] = f2bf(id.w * jd.w);

#pragma unroll
        for (int t = 0; t < 10; ++t) {                   // 10 N-tiles of H
            union { uint4 q[2]; v16bf v; } bfr;
            const uint4* bsrc = (const uint4*)(cur + (t * 16 + nn) * 32 + half * 8);
            bfr.q[0] = bsrc[0]; bfr.q[1] = bsrc[2];
            acc[t] = __builtin_amdgcn_wmma_f32_16x16x32_bf16(
                false, af.v, false, bfr.v, (short)0, acc[t], false, false);
        }
    }

    // prefetch layer-2 chunk 0 into sB[0] (free since chunk 19's barrier); overlaps epilogue
    stage_chunk(W2t, HP, 0, sB[0], tid);

    // ---- epilogue 1: h1 = relu(cross + pre_i + pre_j + de + b1) -> bf16 LDS tile ----
#pragma unroll
    for (int t = 0; t < 10; ++t) {
#pragma unroll
        for (int v = 0; v < 8; ++v) {
            int mm = v + half * 8;
            int k  = w * 16 + mm;
            int h  = t * 16 + nn;
            float val = acc[t][v] + s_pre_i[h]
                      + pre_j[(size_t)s_j[k] * HP + h]
                      + s_de[s_bin[k] * HP + h];
            val = val > 0.f ? val : 0.f;
            sH1[k * HP + h] = f2bf(val);
        }
    }
#pragma unroll
    for (int t = 0; t < 10; ++t) acc[t] = vzero;

    // ========== layer 2: h2 = relu(h1 x W2 + b2)  (K = 160, 5 chunks, dbl-buffered) ==========
    for (int c = 0; c < 5; ++c) {
        const unsigned short* cur = sB[c & 1];
        wait_async0();
        __syncthreads();            // first iter also makes sH1 visible to all waves
        if (c + 1 < 5) stage_chunk(W2t, HP, c + 1, sB[(c + 1) & 1], tid);

        union { uint4 q[2]; v16bf v; } af;
        const unsigned short* arow = sH1 + krow * HP + c * 32 + base;
        af.q[0] = *(const uint4*)(arow);
        af.q[1] = *(const uint4*)(arow + 16);

#pragma unroll
        for (int t = 0; t < 10; ++t) {
            union { uint4 q[2]; v16bf v; } bfr;
            const uint4* bsrc = (const uint4*)(cur + (t * 16 + nn) * 32 + half * 8);
            bfr.q[0] = bsrc[0]; bfr.q[1] = bsrc[2];
            acc[t] = __builtin_amdgcn_wmma_f32_16x16x32_bf16(
                false, af.v, false, bfr.v, (short)0, acc[t], false, false);
        }
    }

    // ========== layer 3: s = relu(h2+b2)·W3 + b3 (fused VALU + shfl reduce) ==========
    float b3v = b3[0];
    float w3v[10], b2v[10];
#pragma unroll
    for (int t = 0; t < 10; ++t) {
        w3v[t] = s_w3[t * 16 + nn];
        b2v[t] = s_b2[t * 16 + nn];
    }
#pragma unroll
    for (int v = 0; v < 8; ++v) {
        float p = 0.f;
#pragma unroll
        for (int t = 0; t < 10; ++t) {
            float h2 = acc[t][v] + b2v[t];
            h2 = h2 > 0.f ? h2 : 0.f;
            p += h2 * w3v[t];
        }
        p += __shfl_xor(p, 1, 32);
        p += __shfl_xor(p, 2, 32);
        p += __shfl_xor(p, 4, 32);
        p += __shfl_xor(p, 8, 32);          // sum over 16-lane half (the H dim)
        if (nn == 0) {
            int k = w * 16 + v + half * 8;
            float sij = s_valid[k] ? (s_sibase[k] + p + b3v) : 0.f;
            out[(size_t)n * (K_ + 1) + k] = sij;
        }
    }
    if (tid == 0) out[(size_t)n * (K_ + 1) + K_] = 0.f;   // eps column
}

// ---------------- host launch ----------------
extern "C" void kernel_launch(void* const* d_in, const int* in_sizes, int n_in,
                              void* d_out, int out_size, void* d_ws, size_t ws_size,
                              hipStream_t stream) {
    const float* g        = (const float*)d_in[0];
    const float* si       = (const float*)d_in[1];
    const int*   i1       = (const int*)d_in[2];
    const int*   i2       = (const int*)d_in[3];
    const float* dist_emb = (const float*)d_in[4];
    const float* W1       = (const float*)d_in[5];
    const float* b1       = (const float*)d_in[6];
    const float* W2       = (const float*)d_in[7];
    const float* b2       = (const float*)d_in[8];
    const float* W3       = (const float*)d_in[9];
    const float* b3       = (const float*)d_in[10];
    float* out = (float*)d_out;

    char* ws = (char*)d_ws;
    float*          gpad  = (float*)(ws + 0);                   // 2,621,440 B
    float*          prei  = (float*)(ws + 2621440);             //   655,360 B
    float*          prej  = (float*)(ws + 3276800);             //   655,360 B
    float*          depre = (float*)(ws + 3932160);             //     5,760 B
    unsigned short* W1ct  = (unsigned short*)(ws + 3938048);    //   204,800 B
    unsigned short* W2t   = (unsigned short*)(ws + 4142848);    //    51,200 B
    float*          b2pad = (float*)(ws + 4194048);             //       640 B
    float*          w3pad = (float*)(ws + 4194688);             //       640 B

    k_gpad<<<(N_ * GP + 255) / 256, 256, 0, stream>>>(g, gpad);
    k_w1ct<<<(HP * GP + 255) / 256, 256, 0, stream>>>(W1, W1ct);
    k_w2t <<<(HP * HP + 255) / 256, 256, 0, stream>>>(W2, W2t);
    k_vec <<<1, 256, 0, stream>>>(b2, W3, b2pad, w3pad);
    k_pre <<<(N_ * HP + 255) / 256, 256, 0, stream>>>(g, W1, b1, prei, prej);
    k_de  <<<(9 * HP + 255) / 256, 256, 0, stream>>>(dist_emb, W1, depre);
    k_main<<<N_, 256, 0, stream>>>(gpad, prei, prej, depre, W1ct, W2t,
                                   b2pad, w3pad, si, i1, i2, b3, out);
}
